// HGConv_22204980920653
// MI455X (gfx1250) — matrix operand order repaired and checked by reading, not compiled, not run on hardware
//
#include <hip/hip_runtime.h>
#include <math.h>

typedef __bf16 bf16;
typedef __attribute__((ext_vector_type(16))) __bf16 v16bf;
typedef __attribute__((ext_vector_type(8)))  float  v8f;

union FragAB { v16bf v; uint4 q[2]; };
union FragC  { v8f   v; float f[8]; };

__device__ inline v8f wmma_bf16(v16bf a, v16bf b, v8f c) {
  // D = A(16x32) * B(32x16) + C, f32 accum
  return __builtin_amdgcn_wmma_f32_16x16x32_bf16(false, a, false, b, (short)0, c, false, false);
}

// ---- DPP16 lane reductions (VALU-only, no DS traffic) ----------------------
// Network: quad_perm(1,0,3,2) -> quad_perm(2,3,0,1) -> row_half_mirror ->
// row_mirror leaves every lane of a 16-lane group holding the group reduction.
// dpp_ctrl must be an ICE at the builtin call site -> template parameter.
template<int CTRL>
__device__ inline float dpp_step(float x) {
  int t = __builtin_amdgcn_update_dpp(0, __float_as_int(x), CTRL, 0xF, 0xF, true);
  return __int_as_float(t);
}
__device__ inline float dpp_max16(float x) {
  x = fmaxf(x, dpp_step<0xB1>(x));   // quad_perm(1,0,3,2)
  x = fmaxf(x, dpp_step<0x4E>(x));   // quad_perm(2,3,0,1)
  x = fmaxf(x, dpp_step<0x141>(x));  // row_half_mirror
  x = fmaxf(x, dpp_step<0x140>(x));  // row_mirror
  return x;
}
__device__ inline float dpp_sum16(float x) {
  x += dpp_step<0xB1>(x);
  x += dpp_step<0x4E>(x);
  x += dpp_step<0x141>(x);
  x += dpp_step<0x140>(x);
  return x;
}

// A-matrix 16x32 bf16 fragment (ISA 7.12.2 layout):
// lane L: row = L%16, K-base = (L<16 ? 0 : 8); halves [kb..kb+7] then [kb+16..kb+23]
__device__ inline v16bf load_frag_a_g(const bf16* base, int ld, int lane) {
  FragAB f;
  const bf16* p = base + (size_t)(lane & 15) * ld + ((lane & 16) >> 1);
  f.q[0] = *(const uint4*)(p);
  f.q[1] = *(const uint4*)(p + 16);
  return f.v;
}
// B-matrix 32x16 bf16 fragment, NT form (B row-major (N,K)):
// lane L: col = L%16, K-base = (L<16 ? 0 : 8); halves [kb..kb+7] then [kb+16..kb+23]
__device__ inline v16bf load_frag_b_g(const bf16* base, int ld, int lane) {
  FragAB f;
  const bf16* p = base + (size_t)(lane & 15) * ld + ((lane & 16) >> 1);
  f.q[0] = *(const uint4*)(p);
  f.q[1] = *(const uint4*)(p + 16);
  return f.v;
}

// ---------------------------------------------------------------------------
// Generic NT GEMM: C(M,N) f32 = A(M,K)bf16 * B(N,K)bf16^T [+ bias]
// MODE 0: plain store. MODE 1: QKV scatter epilogue (q scaled, v transposed).
// Block 256 thr = 8 waves; wave tile 32x32 (2x2 WMMA), block tile 128x64.
// Working set is L2-resident (192MB) so fragments load straight from global.
// ---------------------------------------------------------------------------
template<int MODE>
__global__ __launch_bounds__(256)
void gemm_nt(const bf16* __restrict__ A, const bf16* __restrict__ B,
             const float* __restrict__ bias, float* __restrict__ C,
             int M, int N, int K,
             long long sA, long long sB, long long sC,
             bf16* __restrict__ qb, bf16* __restrict__ kb,
             bf16* __restrict__ vtb, int seqlen, float qscale)
{
  A += (size_t)blockIdx.z * sA;
  B += (size_t)blockIdx.z * sB;
  C += (size_t)blockIdx.z * sC;
  const int lane = threadIdx.x & 31;
  const int w    = threadIdx.x >> 5;
  const int m0 = blockIdx.x * 128 + (w >> 1) * 32;
  const int n0 = blockIdx.y * 64  + (w & 1) * 32;

  FragC acc[2][2] = {};
  for (int k = 0; k < K; k += 32) {
    v16bf a0 = load_frag_a_g(A + (size_t)m0        * K + k, K, lane);
    v16bf a1 = load_frag_a_g(A + (size_t)(m0 + 16) * K + k, K, lane);
    v16bf b0 = load_frag_b_g(B + (size_t)n0        * K + k, K, lane);
    v16bf b1 = load_frag_b_g(B + (size_t)(n0 + 16) * K + k, K, lane);
    acc[0][0].v = wmma_bf16(a0, b0, acc[0][0].v);
    acc[0][1].v = wmma_bf16(a0, b1, acc[0][1].v);
    acc[1][0].v = wmma_bf16(a1, b0, acc[1][0].v);
    acc[1][1].v = wmma_bf16(a1, b1, acc[1][1].v);
  }
  const int rofs = (lane & 16) >> 1;   // C layout: lanes>=16 hold rows M+8
  for (int mt = 0; mt < 2; ++mt)
    for (int nt = 0; nt < 2; ++nt)
      for (int i = 0; i < 8; ++i) {
        int row = m0 + mt * 16 + i + rofs;
        int col = n0 + nt * 16 + (lane & 15);
        float v = acc[mt][nt].f[i];
        if (bias) v += bias[col];
        if (MODE == 0) {
          C[(size_t)row * N + col] = v;
        } else {
          int part = col >> 8, head = (col >> 5) & 7, dim = col & 31;
          int bidx = row / seqlen, sidx = row - bidx * seqlen;
          size_t bh = (size_t)(bidx * 8 + head);
          if (part == 0)
            qb[(bh * seqlen + sidx) * 32 + dim] = (bf16)(v * qscale);
          else if (part == 1)
            kb[(bh * seqlen + sidx) * 32 + dim] = (bf16)v;
          else
            vtb[(bh * 32 + dim) * seqlen + sidx] = (bf16)v;   // V stored transposed
        }
      }
}

// ---------------------------------------------------------------------------
// Flash attention: hd=32 == WMMA K, so S-tile = 1 wmma, O-update = 1 wmma/ntile.
// One 16-query tile per wave, key blocks of 32, online softmax via DPP16
// reductions, C->A layout fix for P through per-wave LDS (8KB total).
// Q pre-scaled, V pre-transposed (VT: hd x S).  Out: (b, S, 256) f32.
// ---------------------------------------------------------------------------
__global__ __launch_bounds__(256)
void flash_attn(const bf16* __restrict__ Q, const bf16* __restrict__ Km,
                const bf16* __restrict__ VT, float* __restrict__ Out, int S)
{
  __shared__ __align__(16) bf16 pbuf[8][16 * 32];
  const int lane = threadIdx.x & 31;
  const int w    = threadIdx.x >> 5;
  const int bh   = blockIdx.y;
  const int bidx = bh >> 3, head = bh & 7;
  const int q0   = blockIdx.x * 128 + w * 16;
  const int rofs = (lane & 16) >> 1;

  v16bf qf = load_frag_a_g(Q + ((size_t)bh * S + q0) * 32, 32, lane);
  FragC o0{}, o1{};
  float mrow[8], lrow[8];
  for (int i = 0; i < 8; ++i) { mrow[i] = -3.0e38f; lrow[i] = 0.f; }

  const bf16* Kbase = Km + (size_t)bh * S * 32;
  const bf16* Vbase = VT + (size_t)bh * 32 * S;
  bf16* pl = &pbuf[w][0];

  for (int kk = 0; kk < S; kk += 32) {
    v16bf kf0 = load_frag_b_g(Kbase + (size_t)kk        * 32, 32, lane);
    v16bf kf1 = load_frag_b_g(Kbase + (size_t)(kk + 16) * 32, 32, lane);
    FragC s0{}, s1{};
    s0.v = wmma_bf16(qf, kf0, s0.v);
    s1.v = wmma_bf16(qf, kf1, s1.v);

    for (int i = 0; i < 8; ++i) {
      float mx = dpp_max16(fmaxf(s0.f[i], s1.f[i]));
      float mnew = fmaxf(mrow[i], mx);
      float corr = __expf(mrow[i] - mnew);
      float p0 = __expf(s0.f[i] - mnew);
      float p1 = __expf(s1.f[i] - mnew);
      float rs = dpp_sum16(p0 + p1);
      lrow[i] = lrow[i] * corr + rs;
      mrow[i] = mnew;
      o0.f[i] *= corr; o1.f[i] *= corr;
      int r = i + rofs;
      pl[r * 32 +      (lane & 15)] = (bf16)p0;
      pl[r * 32 + 16 + (lane & 15)] = (bf16)p1;
    }
    asm volatile("" ::: "memory");   // LDS in-order per wave; stop compiler reordering
    FragAB pf;
    const bf16* pp = pl + (size_t)(lane & 15) * 32 + rofs;
    pf.q[0] = *(const uint4*)(pp);
    pf.q[1] = *(const uint4*)(pp + 16);

    FragAB vf0, vf1;
    const bf16* vp0 = Vbase + (size_t)(lane & 15)        * S + kk + rofs;
    vf0.q[0] = *(const uint4*)(vp0);
    vf0.q[1] = *(const uint4*)(vp0 + 16);
    o0.v = wmma_bf16(pf.v, vf0.v, o0.v);
    const bf16* vp1 = Vbase + (size_t)(16 + (lane & 15)) * S + kk + rofs;
    vf1.q[0] = *(const uint4*)(vp1);
    vf1.q[1] = *(const uint4*)(vp1 + 16);
    o1.v = wmma_bf16(pf.v, vf1.v, o1.v);
    asm volatile("" ::: "memory");
  }
  for (int i = 0; i < 8; ++i) {
    int row = q0 + i + rofs;
    float inv = 1.0f / lrow[i];
    size_t base = ((size_t)bidx * S + row) * 256 + head * 32;
    Out[base +      (lane & 15)] = o0.f[i] * inv;
    Out[base + 16 + (lane & 15)] = o1.f[i] * inv;
  }
}

// ---------------------------------------------------------------------------
// Elementwise / reduction helpers
// ---------------------------------------------------------------------------
__global__ void cvt_bf16_kernel(const float* __restrict__ in, bf16* __restrict__ out,
                                long long n) {
  long long i = (long long)blockIdx.x * blockDim.x + threadIdx.x;
  long long stride = (long long)gridDim.x * blockDim.x;
  for (; i < n; i += stride) out[i] = (bf16)in[i];
}

// in f32 (R,C) -> out bf16 (C,R), batched over z
__global__ __launch_bounds__(256)
void transpose_cvt(const float* __restrict__ in, bf16* __restrict__ out, int R, int C) {
  __shared__ float tile[32][33];
  in  += (size_t)blockIdx.z * R * C;
  out += (size_t)blockIdx.z * R * C;
  int c0 = blockIdx.x * 32, r0 = blockIdx.y * 32;
  for (int j = threadIdx.y; j < 32; j += 8)
    tile[j][threadIdx.x] = in[(size_t)(r0 + j) * C + c0 + threadIdx.x];
  __syncthreads();
  for (int j = threadIdx.y; j < 32; j += 8)
    out[(size_t)(c0 + j) * R + r0 + threadIdx.x] = (bf16)tile[threadIdx.x][j];
}

// softmax over edge axis (rows) per (batch, column), then s *= softmax(s) in place
__global__ __launch_bounds__(256)
void col_softmax_mul(float* __restrict__ sc, int E_, int D_) {
  int col = threadIdx.x;
  float* p = sc + (size_t)blockIdx.x * E_ * D_;
  float mx = -3.0e38f;
  for (int r = 0; r < E_; ++r) mx = fmaxf(mx, p[(size_t)r * D_ + col]);
  float sum = 0.f;
  for (int r = 0; r < E_; ++r) sum += __expf(p[(size_t)r * D_ + col] - mx);
  float inv = 1.0f / sum;
  for (int r = 0; r < E_; ++r) {
    size_t idx = (size_t)r * D_ + col;
    float s = p[idx];
    p[idx] = s * __expf(s - mx) * inv;
  }
}

// LayerNorm(x)*g+b, blend: out = alpha*ef + (1-alpha)*ln + ef   (bf16 out)
__global__ __launch_bounds__(256)
void ln_blend(const float* __restrict__ x, const float* __restrict__ ef,
              const float* __restrict__ g, const float* __restrict__ bta,
              const float* __restrict__ alpha_p, bf16* __restrict__ outbf) {
  const int lane = threadIdx.x & 31;
  const int w    = threadIdx.x >> 5;
  size_t row = (size_t)blockIdx.x * 8 + w;
  const float* xr = x  + row * 256;
  const float* er = ef + row * 256;
  bf16* orow = outbf + row * 256;
  float v[8], s = 0.f;
  for (int j = 0; j < 8; ++j) { v[j] = xr[lane + 32 * j]; s += v[j]; }
  for (int o = 16; o; o >>= 1) s += __shfl_xor(s, o, 32);
  float mu = s * (1.0f / 256.0f);
  float vs = 0.f;
  for (int j = 0; j < 8; ++j) { float d = v[j] - mu; vs += d * d; }
  for (int o = 16; o; o >>= 1) vs += __shfl_xor(vs, o, 32);
  float rstd = rsqrtf(vs * (1.0f / 256.0f) + 1e-5f);
  float alpha = *alpha_p;
  for (int j = 0; j < 8; ++j) {
    int c = lane + 32 * j;
    float y = (v[j] - mu) * rstd * g[c] + bta[c];
    float r = alpha * er[c] + (1.0f - alpha) * y + er[c];
    orow[c] = (bf16)r;
  }
}

__global__ __launch_bounds__(256)
void pool_sum(const float* __restrict__ x, float* __restrict__ out, int E_, int D_) {
  int col = threadIdx.x;
  const float* p = x + (size_t)blockIdx.x * E_ * D_;
  float s = 0.f;
  for (int r = 0; r < E_; ++r) s += p[(size_t)r * D_ + col];
  out[blockIdx.x * D_ + col] = s;
}

// pooled(4,256) @ ec_proj^T -> BN -> @ fc^T + b -> logits(4,16)
__global__ __launch_bounds__(256)
void final_head(const float* __restrict__ pooled, const float* __restrict__ projw,
                const float* __restrict__ bn_g, const float* __restrict__ bn_b,
                const float* __restrict__ bn_mean, const float* __restrict__ bn_var,
                const float* __restrict__ fcw, const float* __restrict__ fcb,
                float* __restrict__ out) {
  __shared__ float bn[4 * 256];
  int t = threadIdx.x;
  for (int bi = 0; bi < 4; ++bi) {
    float acc = 0.f;
    const float* wrow = projw + (size_t)t * 256;
    for (int k = 0; k < 256; ++k) acc += pooled[bi * 256 + k] * wrow[k];
    bn[bi * 256 + t] = (acc - bn_mean[t]) * rsqrtf(bn_var[t] + 1e-5f) * bn_g[t] + bn_b[t];
  }
  __syncthreads();
  if (t < 64) {
    int bi = t >> 4, c = t & 15;
    float acc = fcb[c];
    const float* wrow = fcw + (size_t)c * 256;
    for (int k = 0; k < 256; ++k) acc += bn[bi * 256 + k] * wrow[k];
    out[bi * 16 + c] = acc;
  }
}

// ---------------------------------------------------------------------------
extern "C" void kernel_launch(void* const* d_in, const int* in_sizes, int n_in,
                              void* d_out, int out_size, void* d_ws, size_t ws_size,
                              hipStream_t stream) {
  (void)in_sizes; (void)n_in; (void)out_size; (void)ws_size;
  const float* node      = (const float*)d_in[0];
  const float* edgef     = (const float*)d_in[1];
  const float* inc       = (const float*)d_in[2];
  const float* vc_in_w   = (const float*)d_in[3];
  const float* vc_in_b   = (const float*)d_in[4];
  const float* vc_out_w  = (const float*)d_in[5];
  const float* vc_out_b  = (const float*)d_in[6];
  const float* vc_proj_w = (const float*)d_in[7];
  const float* vc_alpha  = (const float*)d_in[8];
  const float* vc_ln_g   = (const float*)d_in[9];
  const float* vc_ln_b   = (const float*)d_in[10];
  const float* ec_in_w   = (const float*)d_in[11];
  const float* ec_in_b   = (const float*)d_in[12];
  const float* ec_out_w  = (const float*)d_in[13];
  const float* ec_out_b  = (const float*)d_in[14];
  const float* ec_proj_w = (const float*)d_in[15];
  const float* ec_bn_g   = (const float*)d_in[16];
  const float* ec_bn_b   = (const float*)d_in[17];
  const float* ec_bn_mean= (const float*)d_in[18];
  const float* ec_bn_var = (const float*)d_in[19];
  const float* fc_w      = (const float*)d_in[20];
  const float* fc_b      = (const float*)d_in[21];
  float* out = (float*)d_out;

  const int Bn = 4, Mn = 2048, En = 1024, Dn = 256;
  const float qscale = 0.17677669529663687f; // 1/sqrt(32)

  char* ws = (char*)d_ws;
  size_t off = 0;
  auto alloc = [&](size_t bytes) -> void* {
    off = (off + 255) & ~(size_t)255;
    void* p = ws + off;
    off += bytes;
    return p;
  };
  auto nbf = [](size_t n) { return n * sizeof(bf16); };
  auto nf  = [](size_t n) { return n * sizeof(float); };

  bf16* xbf      = (bf16*)alloc(nbf((size_t)Bn * Mn * Dn));
  bf16* w_vin    = (bf16*)alloc(nbf(768 * 256));
  bf16* w_vout   = (bf16*)alloc(nbf(256 * 256));
  bf16* w_vproj  = (bf16*)alloc(nbf(256 * 256));
  bf16* w_ein    = (bf16*)alloc(nbf(768 * 256));
  bf16* w_eout   = (bf16*)alloc(nbf(256 * 256));
  bf16* q1       = (bf16*)alloc(nbf((size_t)Bn * 8 * Mn * 32));
  bf16* k1       = (bf16*)alloc(nbf((size_t)Bn * 8 * Mn * 32));
  bf16* vt1      = (bf16*)alloc(nbf((size_t)Bn * 8 * Mn * 32));
  float* attnA   = (float*)alloc(nf((size_t)Bn * Mn * Dn));
  bf16* attnA_bf = (bf16*)alloc(nbf((size_t)Bn * Mn * Dn));
  float* attn1   = (float*)alloc(nf((size_t)Bn * Mn * Dn));
  bf16* attn1T   = (bf16*)alloc(nbf((size_t)Bn * Dn * Mn));
  bf16* incT     = (bf16*)alloc(nbf((size_t)Bn * En * Mn));
  float* scores  = (float*)alloc(nf((size_t)Bn * En * Dn));
  bf16* edge_bf  = (bf16*)alloc(nbf((size_t)Bn * En * Dn));
  float* eproj   = (float*)alloc(nf((size_t)Bn * En * Dn));
  bf16* edge2_bf = (bf16*)alloc(nbf((size_t)Bn * En * Dn));
  bf16* q2       = (bf16*)alloc(nbf((size_t)Bn * 8 * En * 32));
  bf16* k2       = (bf16*)alloc(nbf((size_t)Bn * 8 * En * 32));
  bf16* vt2      = (bf16*)alloc(nbf((size_t)Bn * 8 * En * 32));
  float* attnB   = (float*)alloc(nf((size_t)Bn * En * Dn));
  bf16* attnB_bf = (bf16*)alloc(nbf((size_t)Bn * En * Dn));
  float* eattn   = (float*)alloc(nf((size_t)Bn * En * Dn));
  float* pooled  = (float*)alloc(nf((size_t)Bn * Dn));

  // --- precision conversion ---
  cvt_bf16_kernel<<<1024, 256, 0, stream>>>(node, xbf, (long long)Bn * Mn * Dn);
  cvt_bf16_kernel<<<192, 256, 0, stream>>>(vc_in_w,  w_vin,  768 * 256);
  cvt_bf16_kernel<<<64,  256, 0, stream>>>(vc_out_w, w_vout, 256 * 256);
  cvt_bf16_kernel<<<64,  256, 0, stream>>>(vc_proj_w,w_vproj,256 * 256);
  cvt_bf16_kernel<<<192, 256, 0, stream>>>(ec_in_w,  w_ein,  768 * 256);
  cvt_bf16_kernel<<<64,  256, 0, stream>>>(ec_out_w, w_eout, 256 * 256);

  // --- MHA #1 on node_feats (s=2048) ---
  gemm_nt<1><<<dim3(8192 / 128, 768 / 64, 1), 256, 0, stream>>>(
      xbf, w_vin, vc_in_b, nullptr, 8192, 768, 256, 0, 0, 0,
      q1, k1, vt1, Mn, qscale);
  flash_attn<<<dim3(Mn / 128, 32), 256, 0, stream>>>(q1, k1, vt1, attnA, Mn);
  cvt_bf16_kernel<<<1024, 256, 0, stream>>>(attnA, attnA_bf, (long long)Bn * Mn * Dn);
  gemm_nt<0><<<dim3(8192 / 128, 256 / 64, 1), 256, 0, stream>>>(
      attnA_bf, w_vout, vc_out_b, attn1, 8192, 256, 256, 0, 0, 0,
      nullptr, nullptr, nullptr, 1, 1.0f);

  // --- incidence aggregation: scores[b] = inc[b]^T @ attn1[b] (NT via transposes) ---
  transpose_cvt<<<dim3(Dn / 32, Mn / 32, Bn), dim3(32, 8), 0, stream>>>(attn1, attn1T, Mn, Dn);
  transpose_cvt<<<dim3(En / 32, Mn / 32, Bn), dim3(32, 8), 0, stream>>>(inc,   incT,   Mn, En);
  gemm_nt<0><<<dim3(En / 128, Dn / 64, Bn), 256, 0, stream>>>(
      incT, attn1T, nullptr, scores, En, Dn, Mn,
      (long long)En * Mn, (long long)Dn * Mn, (long long)En * Dn,
      nullptr, nullptr, nullptr, 1, 1.0f);

  // --- softmax over edges (axis=1) * scores, proj, LN + blend ---
  col_softmax_mul<<<Bn, 256, 0, stream>>>(scores, En, Dn);
  cvt_bf16_kernel<<<512, 256, 0, stream>>>(scores, edge_bf, (long long)Bn * En * Dn);
  gemm_nt<0><<<dim3(4096 / 128, 256 / 64, 1), 256, 0, stream>>>(
      edge_bf, w_vproj, nullptr, eproj, 4096, 256, 256, 0, 0, 0,
      nullptr, nullptr, nullptr, 1, 1.0f);
  ln_blend<<<(Bn * En) / 8, 256, 0, stream>>>(eproj, edgef, vc_ln_g, vc_ln_b,
                                              vc_alpha, edge2_bf);

  // --- MHA #2 on edges (s=1024) ---
  gemm_nt<1><<<dim3(4096 / 128, 768 / 64, 1), 256, 0, stream>>>(
      edge2_bf, w_ein, ec_in_b, nullptr, 4096, 768, 256, 0, 0, 0,
      q2, k2, vt2, En, qscale);
  flash_attn<<<dim3(En / 128, 32), 256, 0, stream>>>(q2, k2, vt2, attnB, En);
  cvt_bf16_kernel<<<512, 256, 0, stream>>>(attnB, attnB_bf, (long long)Bn * En * Dn);
  gemm_nt<0><<<dim3(4096 / 128, 256 / 64, 1), 256, 0, stream>>>(
      attnB_bf, w_eout, ec_out_b, eattn, 4096, 256, 256, 0, 0, 0,
      nullptr, nullptr, nullptr, 1, 1.0f);

  // --- pool, BN, classifier ---
  pool_sum<<<Bn, 256, 0, stream>>>(eattn, pooled, En, Dn);
  final_head<<<1, 256, 0, stream>>>(pooled, ec_proj_w, ec_bn_g, ec_bn_b,
                                    ec_bn_mean, ec_bn_var, fc_w, fc_b, out);
}
